// TransformerEncoder_71597104824390
// MI455X (gfx1250) — matrix operand order repaired and checked by reading, not compiled, not run on hardware
//
#include <hip/hip_runtime.h>
#include <hip/hip_bf16.h>

typedef __attribute__((ext_vector_type(16))) _Float16 v16h;
typedef __attribute__((ext_vector_type(8)))  _Float16 v8h;
typedef __attribute__((ext_vector_type(8)))  float    v8f;

#define WMMA_F16(A, B, C) \
  __builtin_amdgcn_wmma_f32_16x16x32_f16(false, (A), false, (B), (short)0, (C), false, false)

// ---- CDNA5 async global->LDS copy (ASYNCcnt-tracked DMA, no VGPR data path)
__device__ __forceinline__ void async_b128(void* lds, const void* gptr) {
  uint32_t l = (uint32_t)(uintptr_t)lds;   // flat LDS addr[31:0] == LDS byte offset
  asm volatile("global_load_async_to_lds_b128 %0, %1, off"
               :: "v"(l), "v"(gptr) : "memory");
}
#define WAIT_ASYNC_0 asm volatile("s_wait_asynccnt 0" ::: "memory")
#define WAIT_ASYNC_2 asm volatile("s_wait_asynccnt 2" ::: "memory")
#define WAIT_ASYNC_8 asm volatile("s_wait_asynccnt 8" ::: "memory")

// ---- A-fragment gather (contiguous k = 16*half + s, not AMD table order; B loads match): lane<16 -> K{kb..kb+15}, kb=0; lane>=16 kb=16
__device__ __forceinline__ v16h afrag(const _Float16* p) {
  v8h a = *(const v8h*)p;
  v8h b = *(const v8h*)(p + 8);
  return __builtin_shufflevector(a, b, 0,1,2,3,4,5,6,7,8,9,10,11,12,13,14,15);
}

// ---- row-of-16 butterfly reductions (v_permlane16_b32 if available, else ds_bpermute shuffles)
#if __has_builtin(__builtin_amdgcn_permlane16)
__device__ __forceinline__ float pl16(float x, int s0, int s1) {
  int i = __builtin_bit_cast(int, x);
  return __builtin_bit_cast(float, __builtin_amdgcn_permlane16(i, i, s0, s1, false, false));
}
__device__ __forceinline__ float red16_max(float v) {
  v = fmaxf(v, pl16(v, (int)0x67452301, (int)0xEFCDAB89));
  v = fmaxf(v, pl16(v, (int)0x54761032, (int)0xDCFE98BA));
  v = fmaxf(v, pl16(v, (int)0x32107654, (int)0xBA98FEDC));
  v = fmaxf(v, pl16(v, (int)0xFEDCBA98, (int)0x76543210));
  return v;
}
__device__ __forceinline__ float red16_sum(float v) {
  v += pl16(v, (int)0x67452301, (int)0xEFCDAB89);
  v += pl16(v, (int)0x54761032, (int)0xDCFE98BA);
  v += pl16(v, (int)0x32107654, (int)0xBA98FEDC);
  v += pl16(v, (int)0xFEDCBA98, (int)0x76543210);
  return v;
}
#else
__device__ __forceinline__ float red16_max(float v) {
  v = fmaxf(v, __shfl_xor(v, 1, 32)); v = fmaxf(v, __shfl_xor(v, 2, 32));
  v = fmaxf(v, __shfl_xor(v, 4, 32)); v = fmaxf(v, __shfl_xor(v, 8, 32));
  return v;
}
__device__ __forceinline__ float red16_sum(float v) {
  v += __shfl_xor(v, 1, 32); v += __shfl_xor(v, 2, 32);
  v += __shfl_xor(v, 4, 32); v += __shfl_xor(v, 8, 32);
  return v;
}
#endif

// -------------------------------------------------------------------------
// Prologue: f32 -> f16 convert (vectorized)
// -------------------------------------------------------------------------
__global__ __launch_bounds__(256) void cvt_f16_kernel(
    const float* __restrict__ in, _Float16* __restrict__ outp)
{
  const int i = (blockIdx.x * 256 + threadIdx.x) * 8;
  v8h o;
  #pragma unroll
  for (int j = 0; j < 8; ++j) o[j] = (_Float16)in[i + j];
  *(v8h*)&outp[i] = o;
}

// -------------------------------------------------------------------------
// Prologue: W[K,N] f32 -> Wt[N,K] f16 (tiled transpose via LDS)
// -------------------------------------------------------------------------
__global__ __launch_bounds__(256) void transpose_f16_kernel(
    const float* __restrict__ in, _Float16* __restrict__ outp, int K, int N)
{
  __shared__ _Float16 T[32][33];
  const int k0 = blockIdx.x * 32, n0 = blockIdx.y * 32;
  const int r = threadIdx.x >> 3, c = (threadIdx.x & 7) * 4;
  const float* src = in + (size_t)(k0 + r) * N + n0 + c;
  #pragma unroll
  for (int i = 0; i < 4; ++i) T[r][c + i] = (_Float16)src[i];
  __syncthreads();
  _Float16* dst = outp + (size_t)(n0 + r) * K + k0 + c;
  #pragma unroll
  for (int i = 0; i < 4; ++i) dst[i] = T[c + i][r];
}

// -------------------------------------------------------------------------
// C[M,N] = A[M,K] @ Bt[N,K]^T, f16 operands, f32 accum.
// Block 128x128, 8 waves (4M x 2N), wave 32x64, K-step 64,
// double-buffered async global->LDS staging.
// MODE: 0 = f32 out, 1 = f32 out + bias, 2 = f16 out, 3 = f16 out + Vt scatter (qkv)
// -------------------------------------------------------------------------
template<int MODE>
__global__ __launch_bounds__(256) void gemm_f16_kernel(
    const _Float16* __restrict__ A, const _Float16* __restrict__ Bt,
    const float* __restrict__ bias,
    float* __restrict__ C32, _Float16* __restrict__ C16,
    _Float16* __restrict__ Vt,
    int N, int K)
{
  __shared__ alignas(128) _Float16 LA[2][128][64];
  __shared__ alignas(128) _Float16 LB[2][128][64];

  const int tid  = threadIdx.x;
  const int wave = tid >> 5;
  const int lane = tid & 31;
  const int wm   = wave & 3;
  const int wn   = wave >> 2;
  const int m0   = blockIdx.x * 128;
  const int n0   = blockIdx.y * 128;
  const int half = lane >> 4;
  const int l16  = lane & 15;
  const int kb   = half * 16;

  v8f acc[2][4];
  #pragma unroll
  for (int i = 0; i < 2; ++i)
    #pragma unroll
    for (int j = 0; j < 4; ++j)
      #pragma unroll
      for (int r = 0; r < 8; ++r) acc[i][j][r] = 0.0f;

  const int srow = tid >> 1;            // 0..127
  const int soff = (tid & 1) * 32;      // halfs

  auto stage = [&](int buf, int k0) {
    const _Float16* ga = A  + (size_t)(m0 + srow) * K + k0 + soff;
    const _Float16* gb = Bt + (size_t)(n0 + srow) * K + k0 + soff;
    #pragma unroll
    for (int c = 0; c < 4; ++c) {
      async_b128(&LA[buf][srow][soff + c * 8], ga + c * 8);
      async_b128(&LB[buf][srow][soff + c * 8], gb + c * 8);
    }
  };

  const int nk = K >> 6;
  stage(0, 0);
  for (int kt = 0; kt < nk; ++kt) {
    if (kt + 1 < nk) { stage((kt + 1) & 1, (kt + 1) << 6); WAIT_ASYNC_8; }
    else             { WAIT_ASYNC_0; }
    __syncthreads();

    const _Float16 (*la)[64] = LA[kt & 1];
    const _Float16 (*lb)[64] = LB[kt & 1];
    #pragma unroll
    for (int ks = 0; ks < 2; ++ks) {
      v16h af[2], bf[4];
      #pragma unroll
      for (int mt = 0; mt < 2; ++mt)
        af[mt] = afrag(&la[wm * 32 + mt * 16 + l16][ks * 32 + kb]);
      #pragma unroll
      for (int nt = 0; nt < 4; ++nt)
        bf[nt] = *(const v16h*)&lb[wn * 64 + nt * 16 + l16][ks * 32 + half * 16];
      #pragma unroll
      for (int nt = 0; nt < 4; ++nt)
        #pragma unroll
        for (int mt = 0; mt < 2; ++mt)
          acc[mt][nt] = WMMA_F16(af[mt], bf[nt], acc[mt][nt]);
    }
    __syncthreads();
  }

  #pragma unroll
  for (int mt = 0; mt < 2; ++mt) {
    #pragma unroll
    for (int nt = 0; nt < 4; ++nt) {
      const int col  = n0 + wn * 64 + nt * 16 + l16;
      const int row0 = m0 + wm * 32 + mt * 16 + half * 8;
      float bv = (MODE == 1) ? bias[col] : 0.0f;
      #pragma unroll
      for (int r = 0; r < 8; ++r) {
        const float v = acc[mt][nt][r] + bv;
        if (MODE <= 1) C32[(size_t)(row0 + r) * N + col] = v;
        else           C16[(size_t)(row0 + r) * N + col] = (_Float16)v;
      }
      if (MODE == 3 && col >= 1024) {   // V block -> Vt[(b*8+h)*64+d][key], key-contiguous
        const int hh = (col - 1024) >> 6, dd = (col - 1024) & 63;
        const int bq = row0 >> 11, nn = row0 & 2047;
        v8h pk;
        #pragma unroll
        for (int r = 0; r < 8; ++r) pk[r] = (_Float16)acc[mt][nt][r];
        *(v8h*)&Vt[(((size_t)bq * 8 + hh) * 64 + dd) * 2048 + nn] = pk;
      }
    }
  }
}

// -------------------------------------------------------------------------
// Flash attention. qkvh: f16 [b*2048,1536]; Vt: f16 [(b*8+h)*64+d][2048].
// Block = (b,h,128 q rows), 8 waves x 16 q rows. out: f16 [b*2048,512].
// Double-buffered async K/V tile staging; scale folded into f32 scores.
// -------------------------------------------------------------------------
__global__ __launch_bounds__(256) void attn_kernel(
    const _Float16* __restrict__ qkvh, const _Float16* __restrict__ Vt,
    _Float16* __restrict__ out)
{
  __shared__ alignas(128) _Float16 LK[2][32][64];   // [key][d]
  __shared__ alignas(128) _Float16 LV[2][64][32];   // [d][key]
  __shared__ alignas(128) _Float16 LP[8][16][32];   // per-wave P staging

  const int tid  = threadIdx.x;
  const int wave = tid >> 5;
  const int lane = tid & 31;
  const int half = lane >> 4;
  const int l16  = lane & 15;
  const int kb   = half * 16;

  const int b  = blockIdx.x >> 3;
  const int h  = blockIdx.x & 7;
  const int q0 = blockIdx.y * 128 + wave * 16;
  const size_t base  = (size_t)b * 2048;
  const size_t vbase = ((size_t)b * 8 + h) * 64 * 2048;

  // Q fragments straight from global f16 (A-layout gather)
  v16h qf[2];
  {
    const _Float16* qp = qkvh + (base + q0 + l16) * 1536 + h * 64;
    qf[0] = afrag(qp + kb);
    qf[1] = afrag(qp + 32 + kb);
  }

  v8f m_run, l_run, o[4];
  #pragma unroll
  for (int r = 0; r < 8; ++r) { m_run[r] = -3.0e38f; l_run[r] = 0.0f; }
  #pragma unroll
  for (int t = 0; t < 4; ++t)
    #pragma unroll
    for (int r = 0; r < 8; ++r) o[t][r] = 0.0f;

  const int krow = tid >> 3, kch = (tid & 7) * 8;   // K tile: 32 x 64
  const int vrow = tid >> 2, vch = (tid & 3) * 8;   // V tile: 64 x 32

  auto stage = [&](int buf, int j0) {
    async_b128(&LK[buf][krow][kch],
               qkvh + (base + j0 + krow) * 1536 + 512 + h * 64 + kch);
    async_b128(&LV[buf][vrow][vch],
               Vt + vbase + (size_t)vrow * 2048 + j0 + vch);
  };

  stage(0, 0);
  for (int kt = 0; kt < 64; ++kt) {
    const int j0 = kt << 5;
    if (kt < 63) { stage((kt + 1) & 1, j0 + 32); WAIT_ASYNC_2; }
    else         { WAIT_ASYNC_0; }
    __syncthreads();
    const _Float16 (*lk)[64] = LK[kt & 1];
    const _Float16 (*lv)[32] = LV[kt & 1];

    // S = Q @ K^T (scale applied on f32 accumulators)
    v8f s[2];
    #pragma unroll
    for (int jt = 0; jt < 2; ++jt) {
      v8f a;
      #pragma unroll
      for (int r = 0; r < 8; ++r) a[r] = 0.0f;
      #pragma unroll
      for (int ks = 0; ks < 2; ++ks) {
        v16h kf = *(const v16h*)&lk[jt * 16 + l16][ks * 32 + half * 16];
        a = WMMA_F16(qf[ks], kf, a);
      }
      #pragma unroll
      for (int r = 0; r < 8; ++r) a[r] *= 0.125f;
      s[jt] = a;
    }

    // online softmax; slot r = row (r + 8*half), 16 cols across lanes
    v8f alpha;
    #pragma unroll
    for (int r = 0; r < 8; ++r) {
      const float c  = red16_max(fmaxf(s[0][r], s[1][r]));
      const float mn = fmaxf(m_run[r], c);
      alpha[r] = __expf(m_run[r] - mn);
      const float p0 = __expf(s[0][r] - mn);
      const float p1 = __expf(s[1][r] - mn);
      s[0][r] = p0; s[1][r] = p1;
      l_run[r] = l_run[r] * alpha[r] + red16_sum(p0 + p1);
      m_run[r] = mn;
    }
    #pragma unroll
    for (int t = 0; t < 4; ++t)
      #pragma unroll
      for (int r = 0; r < 8; ++r) o[t][r] *= alpha[r];

    // P (C-layout) -> LDS -> A-fragment layout (wave-local, LDS in-order)
    #pragma unroll
    for (int r = 0; r < 8; ++r) {
      LP[wave][r + 8 * half][l16]      = (_Float16)s[0][r];
      LP[wave][r + 8 * half][l16 + 16] = (_Float16)s[1][r];
    }
    v16h pf = afrag(&LP[wave][l16][kb]);
    #pragma unroll
    for (int t = 0; t < 4; ++t) {
      v16h vf = *(const v16h*)&lv[t * 16 + l16][half * 16];
      o[t] = WMMA_F16(pf, vf, o[t]);
    }
    __syncthreads();
  }

  #pragma unroll
  for (int t = 0; t < 4; ++t)
    #pragma unroll
    for (int r = 0; r < 8; ++r)
      out[(base + q0 + r + 8 * half) * 512 + h * 64 + t * 16 + l16] =
          (_Float16)(o[t][r] / l_run[r]);
}

// -------------------------------------------------------------------------
// out = LayerNorm(X + Y) * g + be  (row length 512); optional f16 copy
// -------------------------------------------------------------------------
template<bool DUAL>
__global__ __launch_bounds__(256) void add_ln_kernel(
    const float* __restrict__ X, const float* __restrict__ Y,
    const float* __restrict__ g, const float* __restrict__ be,
    float* __restrict__ out32, _Float16* __restrict__ out16)
{
  __shared__ float ws_[8], wq_[8];
  const int row = blockIdx.x;
  const int tid = threadIdx.x;
  const float* x = X + (size_t)row * 512;
  const float* y = Y + (size_t)row * 512;
  const float v0 = x[tid] + y[tid];
  const float v1 = x[tid + 256] + y[tid + 256];
  float s = v0 + v1;
  float q = v0 * v0 + v1 * v1;
  #pragma unroll
  for (int off = 1; off < 32; off <<= 1) {
    s += __shfl_xor(s, off, 32);
    q += __shfl_xor(q, off, 32);
  }
  if ((tid & 31) == 0) { ws_[tid >> 5] = s; wq_[tid >> 5] = q; }
  __syncthreads();
  float S = 0.0f, Q = 0.0f;
  #pragma unroll
  for (int i = 0; i < 8; ++i) { S += ws_[i]; Q += wq_[i]; }
  const float mu  = S * (1.0f / 512.0f);
  const float var = Q * (1.0f / 512.0f) - mu * mu;
  const float rs  = rsqrtf(var + 1e-6f);
  const float r0  = (v0 - mu) * rs * g[tid]       + be[tid];
  const float r1  = (v1 - mu) * rs * g[tid + 256] + be[tid + 256];
  out32[(size_t)row * 512 + tid]       = r0;
  out32[(size_t)row * 512 + tid + 256] = r1;
  if (DUAL) {
    out16[(size_t)row * 512 + tid]       = (_Float16)r0;
    out16[(size_t)row * 512 + tid + 256] = (_Float16)r1;
  }
}

// -------------------------------------------------------------------------
extern "C" void kernel_launch(void* const* d_in, const int* in_sizes, int n_in,
                              void* d_out, int out_size, void* d_ws, size_t ws_size,
                              hipStream_t stream)
{
  (void)in_sizes; (void)n_in; (void)out_size; (void)ws_size;
  const float* x     = (const float*)d_in[0];
  const float* w_qkv = (const float*)d_in[1];
  const float* w_out = (const float*)d_in[2];
  const float* b_out = (const float*)d_in[3];
  const float* w_ff1 = (const float*)d_in[4];
  const float* w_ff2 = (const float*)d_in[5];
  const float* g1    = (const float*)d_in[6];
  const float* be1   = (const float*)d_in[7];
  const float* g2    = (const float*)d_in[8];
  const float* be2   = (const float*)d_in[9];

  // f32 scratch
  float* f32p = (float*)d_ws;
  float* proj = f32p;                       // 4096*512
  float* res1 = proj + (size_t)4096 * 512;  // 4096*512
  float* ff2  = res1 + (size_t)4096 * 512;  // 4096*512
  // f16 scratch
  _Float16* h = (_Float16*)(ff2 + (size_t)4096 * 512);
  _Float16* xh    = h;                            // 4096*512
  _Float16* qkvh  = xh    + (size_t)4096 * 512;   // 4096*1536
  _Float16* Vth   = qkvh  + (size_t)4096 * 1536;  // 2*8*64*2048
  _Float16* attnh = Vth   + (size_t)2 * 8 * 64 * 2048;  // 4096*512
  _Float16* res1h = attnh + (size_t)4096 * 512;   // 4096*512
  _Float16* ff1h  = res1h + (size_t)4096 * 512;   // 4096*2048
  _Float16* wtqkv = ff1h  + (size_t)4096 * 2048;  // 1536*512
  _Float16* wtout = wtqkv + (size_t)1536 * 512;   // 512*512
  _Float16* wtff1 = wtout + (size_t)512 * 512;    // 2048*512
  _Float16* wtff2 = wtff1 + (size_t)2048 * 512;   // 512*2048
  float* outp = (float*)d_out;

  dim3 blk(256, 1, 1);
  // prologue: one-time f16 conversion / weight transposes
  cvt_f16_kernel<<<1024, blk, 0, stream>>>(x, xh);
  transpose_f16_kernel<<<dim3(16, 48), blk, 0, stream>>>(w_qkv, wtqkv, 512, 1536);
  transpose_f16_kernel<<<dim3(16, 16), blk, 0, stream>>>(w_out, wtout, 512, 512);
  transpose_f16_kernel<<<dim3(16, 64), blk, 0, stream>>>(w_ff1, wtff1, 512, 2048);
  transpose_f16_kernel<<<dim3(64, 16), blk, 0, stream>>>(w_ff2, wtff2, 2048, 512);

  gemm_f16_kernel<3><<<dim3(32, 12), blk, 0, stream>>>(xh, wtqkv, nullptr, nullptr, qkvh, Vth, 1536, 512);
  attn_kernel<<<dim3(16, 16), blk, 0, stream>>>(qkvh, Vth, attnh);
  gemm_f16_kernel<1><<<dim3(32, 4),  blk, 0, stream>>>(attnh, wtout, b_out, proj, nullptr, nullptr, 512, 512);
  add_ln_kernel<true ><<<4096, blk, 0, stream>>>(x, proj, g1, be1, res1, res1h);
  gemm_f16_kernel<2><<<dim3(32, 16), blk, 0, stream>>>(res1h, wtff1, nullptr, nullptr, ff1h, nullptr, 2048, 512);
  gemm_f16_kernel<0><<<dim3(32, 4),  blk, 0, stream>>>(ff1h, wtff2, nullptr, ff2, nullptr, nullptr, 512, 2048);
  add_ln_kernel<false><<<4096, blk, 0, stream>>>(res1, ff2, g2, be2, outp, nullptr);
}
